// SelfAttentionHead_3427383902563
// MI455X (gfx1250) — matrix operand order repaired and checked by reading, not compiled, not run on hardware
//
#include <hip/hip_runtime.h>
#include <hip/hip_bf16.h>

// MI455X / gfx1250, wave32, WMMA bf16 flash-attention.
//   Phase 0: convert W -> W^T bf16 ([H][C]) so projection B-fragments are
//            contiguous b128 loads.
//   Phase 1: Q/K/V projections, one wave per 16x128 output stripe,
//            8x v_wmma_f32_16x16x32_bf16 per K-chunk. 1/sqrt(H) folded into Q.
//   Phase 2: causal flash attention, one wave per 16-row q-tile, online
//            softmax, S and O via v_wmma_f32_16x16x32_bf16.

typedef __bf16 bf16_t;
typedef __attribute__((ext_vector_type(16))) __bf16 v16bf;
typedef __attribute__((ext_vector_type(8)))  __bf16 v8bf;
typedef __attribute__((ext_vector_type(8)))  float  v8f;

#define B_ 8
#define T_ 2048
#define C_ 1024
#define H_ 128

// ---- fragment loaders -------------------------------------------------------
// B-fragment (32x16, KxN, bf16): lane holds 16 contiguous k-elements of one
// column; identity packing into v16bf (element i -> vgpr i/2, half i%2).
static __device__ __forceinline__ v16bf ld16(const bf16_t* p) {
  v8bf lo = *(const v8bf*)p;
  v8bf hi = *(const v8bf*)(p + 8);
  v16bf r;
#pragma unroll
  for (int i = 0; i < 8; ++i) { r[i] = lo[i]; r[i + 8] = hi[i]; }
  return r;
}

// A-fragment (16x32, MxK, bf16): per lane, regs 0-3 <- 8 bf16 at k=16*half,
// regs 4-7 <- 8 bf16 at k=16*half+8 (two separate 16B groups).
static __device__ __forceinline__ v16bf ld16_split(const bf16_t* p0, const bf16_t* p1) {
  v8bf lo = *(const v8bf*)p0;
  v8bf hi = *(const v8bf*)p1;
  v16bf r;
#pragma unroll
  for (int i = 0; i < 8; ++i) { r[i] = lo[i]; r[i + 8] = hi[i]; }
  return r;
}

// ---- phase 0: W [C][H] fp32 -> W^T [H][C] bf16 ------------------------------
__global__ __launch_bounds__(256) void wconv_kernel(
    const float* __restrict__ Wq, const float* __restrict__ Wk,
    const float* __restrict__ Wv,
    bf16_t* __restrict__ Wtq, bf16_t* __restrict__ Wtk, bf16_t* __restrict__ Wtv) {
  const int idx = blockIdx.x * 256 + threadIdx.x;  // 0 .. 3*C_*H_-1
  const int p = idx / (C_ * H_);
  const int e = idx % (C_ * H_);
  const int c = e / H_, h = e % H_;                // coalesced read along h
  const float* W = (p == 0) ? Wq : (p == 1) ? Wk : Wv;
  bf16_t* Wt = (p == 0) ? Wtq : (p == 1) ? Wtk : Wtv;
  Wt[(size_t)h * C_ + c] = (bf16_t)W[e];
}

// ---- phase 1: projections ---------------------------------------------------
// One wave = one 16-row x full-H (16x128) output stripe of one projection.
// 3 * (B*T/16) = 3072 waves. x converted fp32->bf16 once per K-chunk.
__global__ __launch_bounds__(256) void proj_kernel(
    const float* __restrict__ x,
    const bf16_t* __restrict__ Wtq, const bf16_t* __restrict__ Wtk,
    const bf16_t* __restrict__ Wtv,
    bf16_t* __restrict__ Qo, bf16_t* __restrict__ Ko, bf16_t* __restrict__ Vto) {
  const int wave = threadIdx.x >> 5;
  const int lane = threadIdx.x & 31;
  const int lm = lane & 15, lh = lane >> 4;

  const int tid  = blockIdx.x * 8 + wave;          // 0 .. 3071
  const int proj = tid >> 10;                      // 1024 row-tiles per proj
  const int row0 = (tid & 1023) * 16;
  const bf16_t* Wt = (proj == 0) ? Wtq : (proj == 1) ? Wtk : Wtv;

  v8f acc[8];
  {
    v8f z = {};
#pragma unroll
    for (int j = 0; j < 8; ++j) acc[j] = z;
  }

#pragma unroll 1
  for (int kc = 0; kc < C_; kc += 32) {
    // A = x[row0+m][kc..kc+31]  (m = lane&15), fp32 -> bf16
    const float* xr = x + (size_t)(row0 + lm) * C_ + kc + lh * 16;
    v8f fa = *(const v8f*)(xr);
    v8f fb = *(const v8f*)(xr + 8);
    v16bf av;
#pragma unroll
    for (int i = 0; i < 8; ++i) { av[i] = (bf16_t)fa[i]; av[i + 8] = (bf16_t)fb[i]; }
    // B_j = W^T[j*16+n][kc + 16*half + 0..15] : contiguous 32B per lane
#pragma unroll
    for (int j = 0; j < 8; ++j) {
      v16bf bv = ld16(Wt + (size_t)(j * 16 + lm) * C_ + kc + lh * 16);
      acc[j] = __builtin_amdgcn_wmma_f32_16x16x32_bf16(false, av, false, bv,
                                                       (short)0, acc[j], false, false);
    }
  }

  // C layout: lane n = lane&15, row m = r + 8*half
  const float qscale = 0.08838834764831845f;       // 1/sqrt(128), folded into Q
  if (proj == 2) {
    // V stored transposed: Vt[b][h][t]
#pragma unroll
    for (int j = 0; j < 8; ++j)
#pragma unroll
      for (int r = 0; r < 8; ++r) {
        const int row = row0 + r + 8 * lh;         // global token row in [B*T)
        const int b = row >> 11, t = row & (T_ - 1);
        Vto[((size_t)b * H_ + j * 16 + lm) * T_ + t] = (bf16_t)acc[j][r];
      }
  } else if (proj == 0) {
#pragma unroll
    for (int j = 0; j < 8; ++j)
#pragma unroll
      for (int r = 0; r < 8; ++r)
        Qo[(size_t)(row0 + r + 8 * lh) * H_ + j * 16 + lm] =
            (bf16_t)(acc[j][r] * qscale);
  } else {
#pragma unroll
    for (int j = 0; j < 8; ++j)
#pragma unroll
      for (int r = 0; r < 8; ++r)
        Ko[(size_t)(row0 + r + 8 * lh) * H_ + j * 16 + lm] = (bf16_t)acc[j][r];
  }
}

// ---- phase 2: causal flash attention ---------------------------------------
// One wave per 16-row q-tile; full H=128 accumulated in 8 v8f fragments.
__global__ __launch_bounds__(256) void attn_kernel(
    const bf16_t* __restrict__ Q, const bf16_t* __restrict__ K,
    const bf16_t* __restrict__ Vt, float* __restrict__ O) {
  __shared__ __align__(16) bf16_t pstage[8][16 * 32];   // P re-layout buffer
  const int wave = threadIdx.x >> 5;
  const int lane = threadIdx.x & 31;
  const int lm = lane & 15, lh = lane >> 4;

  const int tile = blockIdx.x * 8 + wave;    // 0..1023
  const int b  = tile >> 7;                  // 128 q-tiles per batch
  const int t0 = (tile & 127) << 4;
  const bf16_t* Qb = Q  + (size_t)b * T_ * H_;
  const bf16_t* Kb = K  + (size_t)b * T_ * H_;
  const bf16_t* Vb = Vt + (size_t)b * H_ * T_;

  // Resident Q A-fragments, 4 chunks of K=32 covering H=128 (Q pre-scaled)
  v16bf qa[4];
#pragma unroll
  for (int c = 0; c < 4; ++c) {
    const bf16_t* q = Qb + (size_t)(t0 + lm) * H_ + c * 32;
    qa[c] = ld16_split(q + lh * 16, q + lh * 16 + 8);
  }

  v8f acc[8];
  {
    v8f z = {};
#pragma unroll
    for (int j = 0; j < 8; ++j) acc[j] = z;
  }
  float rm[8], rl[8];
#pragma unroll
  for (int r = 0; r < 8; ++r) { rm[r] = -1e30f; rl[r] = 0.f; }

  bf16_t* pp = pstage[wave];
  const int nkb = (t0 + 47) >> 5;            // causal: keys 0..t0+15 only
#pragma unroll 1
  for (int kb = 0; kb < nkb; ++kb) {
    const int n0 = kb << 5;
    // S = Q K^T : two 16x16 tiles (keys n0..n0+15 and n0+16..n0+31)
    v8f s0 = {}, s1 = {};
#pragma unroll
    for (int c = 0; c < 4; ++c) {
      v16bf k0 = ld16(Kb + (size_t)(n0 + lm) * H_ + c * 32 + lh * 16);
      s0 = __builtin_amdgcn_wmma_f32_16x16x32_bf16(false, qa[c], false, k0,
                                                   (short)0, s0, false, false);
      v16bf k1 = ld16(Kb + (size_t)(n0 + 16 + lm) * H_ + c * 32 + lh * 16);
      s1 = __builtin_amdgcn_wmma_f32_16x16x32_bf16(false, qa[c], false, k1,
                                                   (short)0, s1, false, false);
    }
    // causal mask; C layout: row m = r + 8*lh, col n = lm
    const int qrow = t0 + 8 * lh;
    float bm[8];
#pragma unroll
    for (int r = 0; r < 8; ++r) {
      float a  = s0[r];
      float c2 = s1[r];
      if (n0 + lm > qrow + r)      a  = -1e30f;
      if (n0 + 16 + lm > qrow + r) c2 = -1e30f;
      s0[r] = a; s1[r] = c2;
      bm[r] = fmaxf(a, c2);
    }
    // row-max across the 16 lanes holding one row
#pragma unroll
    for (int m = 1; m < 16; m <<= 1)
#pragma unroll
      for (int r = 0; r < 8; ++r)
        bm[r] = fmaxf(bm[r], __shfl_xor(bm[r], m, 32));
    float alpha[8], bs[8];
#pragma unroll
    for (int r = 0; r < 8; ++r) {
      const float mn = fmaxf(rm[r], bm[r]);
      alpha[r] = __expf(rm[r] - mn);
      rm[r] = mn;
      s0[r] = __expf(s0[r] - mn);
      s1[r] = __expf(s1[r] - mn);
      bs[r] = s0[r] + s1[r];
    }
#pragma unroll
    for (int m = 1; m < 16; m <<= 1)
#pragma unroll
      for (int r = 0; r < 8; ++r)
        bs[r] += __shfl_xor(bs[r], m, 32);
#pragma unroll
    for (int r = 0; r < 8; ++r) rl[r] = rl[r] * alpha[r] + bs[r];
#pragma unroll
    for (int j = 0; j < 8; ++j)
#pragma unroll
      for (int r = 0; r < 8; ++r) acc[j][r] *= alpha[r];
    // C-layout P -> row-major [16][32] bf16 in LDS, then reload as A-fragment
#pragma unroll
    for (int r = 0; r < 8; ++r) {
      const int row = r + 8 * lh;
      pp[row * 32 + lm]      = (bf16_t)s0[r];
      pp[row * 32 + 16 + lm] = (bf16_t)s1[r];
    }
    asm volatile("s_wait_dscnt 0" ::: "memory");   // LDS stores visible to reload
    v16bf pa = ld16_split(pp + lm * 32 + lh * 16, pp + lm * 32 + lh * 16 + 8);
    // O += P * V : V^T layout makes B-fragments contiguous 32B loads
#pragma unroll
    for (int j = 0; j < 8; ++j) {
      v16bf vb = ld16(Vb + (size_t)(j * 16 + lm) * T_ + n0 + lh * 16);
      acc[j] = __builtin_amdgcn_wmma_f32_16x16x32_bf16(false, pa, false, vb,
                                                       (short)0, acc[j], false, false);
    }
  }
  // normalize + store fp32 output
#pragma unroll
  for (int r = 0; r < 8; ++r) rl[r] = 1.f / rl[r];
  float* Ob = O + (size_t)b * T_ * H_;
#pragma unroll
  for (int j = 0; j < 8; ++j)
#pragma unroll
    for (int r = 0; r < 8; ++r)
      Ob[(size_t)(t0 + r + 8 * lh) * H_ + j * 16 + lm] = acc[j][r] * rl[r];
}

extern "C" void kernel_launch(void* const* d_in, const int* in_sizes, int n_in,
                              void* d_out, int out_size, void* d_ws, size_t ws_size,
                              hipStream_t stream) {
  (void)in_sizes; (void)n_in; (void)out_size; (void)ws_size;
  const float* x  = (const float*)d_in[0];
  const float* Wk = (const float*)d_in[1];
  const float* Wq = (const float*)d_in[2];
  const float* Wv = (const float*)d_in[3];
  float* out = (float*)d_out;

  bf16_t* Qo  = (bf16_t*)d_ws;                        // [B][T][H] bf16 (pre-scaled)
  bf16_t* Ko  = Qo  + (size_t)B_ * T_ * H_;           // [B][T][H] bf16
  bf16_t* Vto = Ko  + (size_t)B_ * T_ * H_;           // [B][H][T] bf16
  bf16_t* Wtq = Vto + (size_t)B_ * H_ * T_;           // [H][C] bf16
  bf16_t* Wtk = Wtq + (size_t)H_ * C_;
  bf16_t* Wtv = Wtk + (size_t)H_ * C_;

  // phase 0: 3*C*H elements / 256
  wconv_kernel<<<(3 * C_ * H_) / 256, 256, 0, stream>>>(Wq, Wk, Wv, Wtq, Wtk, Wtv);
  // phase 1: 3 * 1024 stripe-waves / 8 per block
  proj_kernel<<<384, 256, 0, stream>>>(x, Wtq, Wtk, Wtv, Qo, Ko, Vto);
  // phase 2: 1024 q-tiles / 8 waves per block
  attn_kernel<<<128, 256, 0, stream>>>(Qo, Ko, Vto, out);
}
